// LSTMClassifier_8615704395958
// MI455X (gfx1250) — 8-bit matrix operand order repaired and checked by reading, compiled, not run on hardware
//
#include <hip/hip_runtime.h>
#include <cstdint>

#define N_NODES 8192
#define FEAT 128
#define EMB 64
#define HID 128
#define G4 512
#define KSTEPS 64
#define MTILES 512
#define NTILES 8

typedef int   v16i __attribute__((ext_vector_type(16)));
typedef float v8f  __attribute__((ext_vector_type(8)));
typedef int   vec4i __attribute__((vector_size(16)));

__device__ __forceinline__ uint32_t rotl32(uint32_t v, int r) {
    return (v << r) | (v >> (32 - r));
}
__device__ __forceinline__ void tf_round4(uint32_t& x0, uint32_t& x1,
                                          int r0, int r1, int r2, int r3) {
    x0 += x1; x1 = rotl32(x1, r0); x1 ^= x0;
    x0 += x1; x1 = rotl32(x1, r1); x1 ^= x0;
    x0 += x1; x1 = rotl32(x1, r2); x1 ^= x0;
    x0 += x1; x1 = rotl32(x1, r3); x1 ^= x0;
}
__device__ __forceinline__ void threefry2x32(uint32_t k0, uint32_t k1,
                                             uint32_t c0, uint32_t c1,
                                             uint32_t& o0, uint32_t& o1) {
    uint32_t ks2 = 0x1BD11BDAu ^ k0 ^ k1;
    uint32_t x0 = c0 + k0, x1 = c1 + k1;
    tf_round4(x0, x1, 13, 15, 26, 6);  x0 += k1;  x1 += ks2 + 1u;
    tf_round4(x0, x1, 17, 29, 16, 24); x0 += ks2; x1 += k0 + 2u;
    tf_round4(x0, x1, 13, 15, 26, 6);  x0 += k0;  x1 += k1 + 3u;
    tf_round4(x0, x1, 17, 29, 16, 24); x0 += k1;  x1 += ks2 + 4u;
    tf_round4(x0, x1, 13, 15, 26, 6);  x0 += ks2; x1 += k0 + 5u;
    o0 = x0; o1 = x1;
}
__device__ __forceinline__ uint32_t split_elem(uint32_t j) {
    uint32_t o0, o1;
    if (j < 8191u) { threefry2x32(0u, 42u, j, 8191u + j, o0, o1); return o0; }
    threefry2x32(0u, 42u, j - 8191u, j, o0, o1); return o1;
}
__device__ __forceinline__ float bits_to_uniform(uint32_t b) {
    float u = __uint_as_float((b >> 9) | 0x3f800000u) - 1.0f;
    return 0.01f + u * 0.09f;
}
__device__ __forceinline__ float sigm(float x) { return 1.0f / (1.0f + __expf(-x)); }

__device__ __forceinline__ void async_cp16(const uint32_t* g, uint32_t* l) {
    __builtin_amdgcn_global_load_async_to_lds_b128(
        (__attribute__((address_space(1))) vec4i*)(uintptr_t)g,
        (__attribute__((address_space(3))) vec4i*)(uintptr_t)l,
        0, 0);
}
__device__ __forceinline__ void wait_async0() {
    asm volatile("s_wait_asynccnt 0x0" ::: "memory");
}

__global__ void k_bfrag(const int* __restrict__ tags, uint32_t* __restrict__ bfrag) {
    int w = blockIdx.x * blockDim.x + threadIdx.x;
    if (w >= KSTEPS * NTILES * 32 * 16) return;
    int v = w & 15, lane = (w >> 4) & 31, nt = (w >> 9) & 7, ks = w >> 12;
    int n  = nt * 16 + (lane & 15);
    int kb = ks * 128 + 32 * (v >> 2) + ((lane >> 4) << 4) + (v & 3) * 4;
    uint32_t word = 0;
#pragma unroll
    for (int b = 0; b < 4; ++b)
        if (tags[kb + b] == n) word |= 0x38u << (8 * b);
    bfrag[w] = word;
}

__global__ void k_adjbits(const float* __restrict__ adj, uint32_t* __restrict__ bits) {
    int w = blockIdx.x * blockDim.x + threadIdx.x;
    if (w >= N_NODES * 256) return;
    const float* p = adj + (size_t)w * 32;
    uint32_t r = 0;
#pragma unroll
    for (int m = 0; m < 32; ++m)
        if (p[m] != 0.0f) r |= (1u << m);
    bits[w] = r;
}

__global__ void k_trans(const float* __restrict__ Wih, const float* __restrict__ Whh,
                        float* __restrict__ WihT, float* __restrict__ WhhT) {
    int idx = blockIdx.x * blockDim.x + threadIdx.x;
    if (idx >= G4 * HID) return;
    int j = idx >> 7, k = idx & 127;
    WihT[k * G4 + j] = Wih[idx];
    WhhT[k * G4 + j] = Whh[idx];
}

__global__ void k_gemm_counts(const float* __restrict__ adj,
                              const uint32_t* __restrict__ bfrag,
                              float* __restrict__ counts) {
    __shared__ uint32_t Bs[2][NTILES * 32 * 16];
    int t    = threadIdx.x;
    int wv   = t >> 5;
    int lane = t & 31;
    int mtile = blockIdx.x * 8 + wv;
    int mrow  = mtile * 16 + (lane & 15);
    int khalf = (lane >> 4) * 8;
    const float* arow = adj + (size_t)mrow * N_NODES;

#pragma unroll
    for (int r = 0; r < 4; ++r)
        async_cp16(bfrag + (size_t)(r * 256 + t) * 4, &Bs[0][(r * 256 + t) * 4]);
    wait_async0();
    __syncthreads();

    v8f acc[NTILES] = {};
    union { uint32_t u[16]; v16i v; } A;
    union { uint32_t u[16]; v16i v; } B;

    for (int ks = 0; ks < KSTEPS; ++ks) {
        int cur = ks & 1, nxt = cur ^ 1;
        if (ks + 1 < KSTEPS) {
            const uint32_t* src = bfrag + (size_t)(ks + 1) * (NTILES * 32 * 16);
#pragma unroll
            for (int r = 0; r < 4; ++r)
                async_cp16(src + (size_t)(r * 256 + t) * 4, &Bs[nxt][(r * 256 + t) * 4]);
        }
        int kbase = ks * 128;
#pragma unroll
        for (int v = 0; v < 16; ++v) {
            int k = kbase + ((v >> 2) * 32) + ((v & 3) * 4) + 2 * khalf;
            float4 f = *(const float4*)(arow + k);
            uint32_t w = 0;
            if (f.x != 0.0f) w |= 0x38u;
            if (f.y != 0.0f) w |= 0x38u << 8;
            if (f.z != 0.0f) w |= 0x38u << 16;
            if (f.w != 0.0f) w |= 0x38u << 24;
            A.u[v] = w;
        }
#pragma unroll
        for (int nt = 0; nt < NTILES; ++nt) {
            const uint32_t* bp = &Bs[cur][(nt * 32 + lane) * 16];
#pragma unroll
            for (int v = 0; v < 16; ++v) B.u[v] = bp[v];
            acc[nt] = __builtin_amdgcn_wmma_f32_16x16x128_fp8_fp8(
                A.v, B.v, (short)0, acc[nt], false, false);
        }
        wait_async0();
        __syncthreads();
    }
    int ncol  = lane & 15;
    int mbase = mtile * 16 + ((lane >> 4) * 8);
#pragma unroll
    for (int nt = 0; nt < NTILES; ++nt)
#pragma unroll
        for (int j = 0; j < 8; ++j)
            counts[(size_t)(mbase + j) * FEAT + nt * 16 + ncol] = acc[nt][j];
}

__global__ void k_feat(const int* __restrict__ tags, const float* __restrict__ counts,
                       const float* __restrict__ Wemb, const float* __restrict__ bemb,
                       float* __restrict__ X) {
    __shared__ float cs[FEAT];
    int n = blockIdx.x, e = threadIdx.x;
    cs[e]      = counts[(size_t)n * FEAT + e];
    cs[64 + e] = counts[(size_t)n * FEAT + 64 + e];
    __syncthreads();
    float be = bemb[e];
    float node = Wemb[tags[n] * EMB + e] + be;
    float acc = be;
    for (int k = 0; k < FEAT; ++k) acc += cs[k] * Wemb[k * EMB + e];
    X[(size_t)n * 128 + e]      = node;
    X[(size_t)n * 128 + 64 + e] = acc;
}

__global__ void k_gpre(const float* __restrict__ X, const float* __restrict__ WihT,
                       const float* __restrict__ bih, const float* __restrict__ bhh,
                       float* __restrict__ G) {
    __shared__ float xs[16][128];
    int nb = blockIdx.x * 16;
    int t = threadIdx.x;
    for (int idx = t; idx < 16 * 128; idx += 512)
        xs[idx >> 7][idx & 127] = X[(size_t)(nb + (idx >> 7)) * 128 + (idx & 127)];
    __syncthreads();
    float acc[16];
    float bias = bih[t] + bhh[t];
#pragma unroll
    for (int r = 0; r < 16; ++r) acc[r] = bias;
    for (int k = 0; k < 128; ++k) {
        float w = WihT[k * G4 + t];
#pragma unroll
        for (int r = 0; r < 16; ++r) acc[r] += xs[r][k] * w;
    }
#pragma unroll
    for (int r = 0; r < 16; ++r) G[(size_t)(nb + r) * G4 + t] = acc[r];
}

__global__ void k_order(const uint32_t* __restrict__ adjbits, int* __restrict__ order) {
    __shared__ uint32_t row[256];
    __shared__ unsigned char visited[N_NODES];
    __shared__ float redv[1024];
    __shared__ int   redi[1024];
    __shared__ uint32_t skey[2];
    __shared__ int scur;
    int t = threadIdx.x;
    for (int m = t; m < N_NODES; m += 1024) visited[m] = 0;
    if (t == 0) { visited[0] = 1; scur = 0; order[0] = 0; }
    __syncthreads();

    for (int s = 1; s < N_NODES; ++s) {
        int i = scur;
        if (t == 0) {
            uint32_t q = (uint32_t)(s - 1);
            skey[0] = split_elem(2u * q);
            skey[1] = split_elem(2u * q + 1u);
        }
        if (t < 256) row[t] = adjbits[(size_t)i * 256 + t];
        __syncthreads();

        uint32_t k0 = skey[0], k1 = skey[1];
        float bestv = -1.0f; int besti = N_NODES;
#pragma unroll
        for (int c = 0; c < 4; ++c) {
            int p = t * 4 + c;
            uint32_t o0, o1;
            threefry2x32(k0, k1, (uint32_t)p, (uint32_t)(4096 + p), o0, o1);
            {
                int m = p;
                float v = visited[m] ? -1.0f
                          : bits_to_uniform(o0) + (((row[m >> 5] >> (m & 31)) & 1u) ? 1.0f : 0.0f);
                if (v > bestv || (v == bestv && m < besti)) { bestv = v; besti = m; }
            }
            {
                int m = p + 4096;
                float v = visited[m] ? -1.0f
                          : bits_to_uniform(o1) + (((row[m >> 5] >> (m & 31)) & 1u) ? 1.0f : 0.0f);
                if (v > bestv || (v == bestv && m < besti)) { bestv = v; besti = m; }
            }
        }
        redv[t] = bestv; redi[t] = besti;
        __syncthreads();
        for (int off = 512; off > 0; off >>= 1) {
            if (t < off) {
                float v2 = redv[t + off]; int i2 = redi[t + off];
                if (v2 > redv[t] || (v2 == redv[t] && i2 < redi[t])) { redv[t] = v2; redi[t] = i2; }
            }
            __syncthreads();
        }
        if (t == 0) {
            int inew = redi[0];
            order[s] = inew; visited[inew] = 1; scur = inew;
        }
        __syncthreads();
    }
}

__global__ void k_lstm(const float* __restrict__ G, const float* __restrict__ WhhT,
                       const int* __restrict__ order, float* __restrict__ embed) {
    __shared__ float h[HID], c[HID], g[G4], emb[HID];
    int t = threadIdx.x;
    if (t < HID) { h[t] = 0.0f; c[t] = 0.0f; emb[t] = 0.0f; }
    __syncthreads();
    for (int s = 0; s < N_NODES; ++s) {
        int n = order[s];
        float acc = G[(size_t)n * G4 + t];
        for (int k = 0; k < HID; ++k) acc += WhhT[k * G4 + t] * h[k];
        g[t] = acc;
        __syncthreads();
        if (t < HID) {
            float gi = g[t], gf = g[HID + t], gc = g[2 * HID + t], go = g[3 * HID + t];
            float cn = sigm(gf) * c[t] + sigm(gi) * tanhf(gc);
            float hn = sigm(go) * tanhf(cn);
            c[t] = cn; h[t] = hn; emb[t] += hn;
        }
        __syncthreads();
    }
    if (t < HID) embed[t] = emb[t];
}

__global__ void k_head(const float* __restrict__ embed,
                       const float* __restrict__ W1, const float* __restrict__ b1,
                       const float* __restrict__ W2, const float* __restrict__ b2,
                       const int* __restrict__ label, float* __restrict__ out) {
    __shared__ float e[HID], h1[HID], logit[8];
    int t = threadIdx.x;
    e[t] = embed[t];
    __syncthreads();
    float a = b1[t];
    for (int k = 0; k < HID; ++k) a += e[k] * W1[k * 128 + t];
    h1[t] = a > 0.0f ? a : 0.0f;
    __syncthreads();
    if (t < 8) {
        float l = b2[t];
        for (int j = 0; j < 128; ++j) l += h1[j] * W2[j * 8 + t];
        logit[t] = l;
    }
    __syncthreads();
    if (t == 0) {
        float m = logit[0];
        for (int ci = 1; ci < 8; ++ci) m = fmaxf(m, logit[ci]);
        float ss = 0.0f;
        for (int ci = 0; ci < 8; ++ci) ss += expf(logit[ci] - m);
        float lse = logf(ss);
        float ls[8];
        for (int ci = 0; ci < 8; ++ci) { ls[ci] = logit[ci] - m - lse; out[ci] = ls[ci]; }
        out[8] = -ls[label[0] & 7];
    }
}

extern "C" void kernel_launch(void* const* d_in, const int* in_sizes, int n_in,
                              void* d_out, int out_size, void* d_ws, size_t ws_size,
                              hipStream_t stream) {
    const int*   tags  = (const int*)d_in[0];
    const float* adj   = (const float*)d_in[1];
    const int*   label = (const int*)d_in[2];
    const float* Wemb  = (const float*)d_in[3];
    const float* bemb  = (const float*)d_in[4];
    const float* Wih   = (const float*)d_in[5];
    const float* Whh   = (const float*)d_in[6];
    const float* bih   = (const float*)d_in[7];
    const float* bhh   = (const float*)d_in[8];
    const float* W1    = (const float*)d_in[9];
    const float* b1    = (const float*)d_in[10];
    const float* W2    = (const float*)d_in[11];
    const float* b2    = (const float*)d_in[12];
    float* out = (float*)d_out;

    char* ws = (char*)d_ws;
    size_t off = 0;
    auto carve = [&](size_t bytes) -> void* {
        void* p = ws + off;
        off = (off + bytes + 255) & ~(size_t)255;
        return p;
    };
    uint32_t* bfrag   = (uint32_t*)carve((size_t)KSTEPS * NTILES * 32 * 64);
    uint32_t* adjbits = (uint32_t*)carve((size_t)N_NODES * 256 * 4);
    float*    counts  = (float*)carve((size_t)N_NODES * FEAT * 4);
    float*    X       = (float*)carve((size_t)N_NODES * 128 * 4);
    float*    G       = (float*)carve((size_t)N_NODES * G4 * 4);
    float*    WihT    = (float*)carve((size_t)G4 * HID * 4);
    float*    WhhT    = (float*)carve((size_t)G4 * HID * 4);
    int*      order   = (int*)carve((size_t)N_NODES * 4);
    float*    embed   = (float*)carve((size_t)HID * 4);

    k_bfrag<<<dim3(1024), dim3(256), 0, stream>>>(tags, bfrag);
    k_adjbits<<<dim3(8192), dim3(256), 0, stream>>>(adj, adjbits);
    k_trans<<<dim3(256), dim3(256), 0, stream>>>(Wih, Whh, WihT, WhhT);
    k_gemm_counts<<<dim3(64), dim3(256), 0, stream>>>(adj, bfrag, counts);
    k_feat<<<dim3(N_NODES), dim3(64), 0, stream>>>(tags, counts, Wemb, bemb, X);
    k_gpre<<<dim3(N_NODES / 16), dim3(512), 0, stream>>>(X, WihT, bih, bhh, G);
    k_order<<<dim3(1), dim3(1024), 0, stream>>>(adjbits, order);
    k_lstm<<<dim3(1), dim3(512), 0, stream>>>(G, WhhT, order, embed);
    k_head<<<dim3(1), dim3(128), 0, stream>>>(embed, W1, b1, W2, b2, label, out);
}
